// net_81183471829206
// MI455X (gfx1250) — matrix operand order repaired and checked by reading, not compiled, not run on hardware
//
#include <hip/hip_runtime.h>
#include <cstddef>

// ---------------- model constants ----------------
#define Hd    200
#define Td    5
#define FTd   40
#define HEADSd 5
#define DHd   40
#define Bd    32
#define NAd   64
#define NRd   512
#define NMd   2048      // B*NA
#define NPRd  16384     // B*NR
#define EMd   8192      // mol edges
#define EPd   98304     // prot edges

typedef __attribute__((ext_vector_type(16))) __bf16         v16bf;
typedef __attribute__((ext_vector_type(16))) unsigned short v16us;
typedef __attribute__((ext_vector_type(8)))  unsigned short v8us;
typedef __attribute__((ext_vector_type(8)))  float          v8f;

__device__ __forceinline__ unsigned short f2bf(float f) {
  unsigned u = __float_as_uint(f);
  return (unsigned short)((u + 0x7FFFu + ((u >> 16) & 1u)) >> 16); // RNE
}
__device__ __forceinline__ v8us cvt8(float4 a, float4 b) {
  v8us h;
  h[0] = f2bf(a.x); h[1] = f2bf(a.y); h[2] = f2bf(a.z); h[3] = f2bf(a.w);
  h[4] = f2bf(b.x); h[5] = f2bf(b.y); h[6] = f2bf(b.z); h[7] = f2bf(b.w);
  return h;
}
// monotone float <-> uint mapping for atomic min/max
__device__ __forceinline__ unsigned fenc(float f) {
  unsigned u = __float_as_uint(f);
  return (u & 0x80000000u) ? ~u : (u | 0x80000000u);
}
__device__ __forceinline__ float fdec(unsigned e) {
  unsigned u = (e & 0x80000000u) ? (e ^ 0x80000000u) : ~e;
  return __uint_as_float(u);
}

// ---------------- WMMA bf16 GEMM: C = act(A[MxK] @ W[KxN] + bias), row-major ----
// 256 threads = 8 wave32 (4x2 wave grid). Block tile 128x64, wave tile 32x32
// (2 A frags x 2 B frags -> 4 v_wmma_f32_16x16x32_bf16 per K-step).
#define TBM 128
#define TBN 64
#define TBK 32
#define LDP 40   // padded LDS row stride (ushorts): 80 B, 16B-aligned, bank-spreading

template <int ACT>
__global__ __launch_bounds__(256) void k_gemm(
    const float* __restrict__ A, const float* __restrict__ W,
    const float* __restrict__ bias, float* __restrict__ C,
    int M, int N, int K, int ldc)
{
  __shared__ alignas(16) unsigned short As[TBM][LDP];  // A tile, row-major (bf16 bits)
  __shared__ alignas(16) unsigned short Bt[TBN][LDP];  // W tile TRANSPOSED: Bt[n][k]
  const int tid   = threadIdx.x;
  const int wave  = tid >> 5;
  const int lane  = tid & 31;
  const int r0    = (wave >> 1) << 5;   // wave rows: 0/32/64/96
  const int c0    = (wave & 1) << 5;    // wave cols: 0/32
  const int bm    = blockIdx.y * TBM;
  const int bn    = blockIdx.x * TBN;
  const int mfrag = lane & 15;          // M (A) / N (B,C) index within 16
  const int klo   = (lane >> 4) << 4;   // A frag: 0 or 16
  const int kbB   = (lane >> 4) << 4;   // B frag: 0 or 16

  v8f acc[2][2];
#pragma unroll
  for (int i = 0; i < 2; ++i)
#pragma unroll
    for (int j = 0; j < 2; ++j) acc[i][j] = (v8f){0.f,0.f,0.f,0.f,0.f,0.f,0.f,0.f};

  const bool mfull = (bm + TBM) <= M;
  const bool nfull = (bn + TBN) <= N;
  const bool avec  = mfull && ((K & 3) == 0);  // 16B-aligned fast path legal
  const bool wvec  = nfull && ((N & 3) == 0);

  for (int k0 = 0; k0 < K; k0 += TBK) {
    const bool kfull = (k0 + TBK) <= K;
    // ---- stage A tile (128x32) ----
    if (avec && kfull) {
      int r  = tid >> 1;           // 0..127
      int ch = (tid & 1) << 4;     // 0 or 16
      const float4* ap = (const float4*)(A + (size_t)(bm + r) * K + k0 + ch);
      float4 f0 = ap[0], f1 = ap[1], f2 = ap[2], f3 = ap[3];
      *(v8us*)&As[r][ch]     = cvt8(f0, f1);
      *(v8us*)&As[r][ch + 8] = cvt8(f2, f3);
    } else {
      for (int i = tid; i < TBM * TBK; i += 256) {
        int r = i >> 5, c = i & 31;
        int gr = bm + r, gc = k0 + c;
        float va = (gr < M && gc < K) ? A[(size_t)gr * K + gc] : 0.f;
        As[r][c] = f2bf(va);
      }
    }
    // ---- stage W tile (32x64) transposed into Bt[n][k] ----
    if (wvec && kfull) {
      int r  = tid >> 3;           // k row 0..31
      int cb = (tid & 7) << 3;     // col base 0..56
      const float4* wp = (const float4*)(W + (size_t)(k0 + r) * N + bn + cb);
      float4 g0 = wp[0], g1 = wp[1];
      Bt[cb + 0][r] = f2bf(g0.x); Bt[cb + 1][r] = f2bf(g0.y);
      Bt[cb + 2][r] = f2bf(g0.z); Bt[cb + 3][r] = f2bf(g0.w);
      Bt[cb + 4][r] = f2bf(g1.x); Bt[cb + 5][r] = f2bf(g1.y);
      Bt[cb + 6][r] = f2bf(g1.z); Bt[cb + 7][r] = f2bf(g1.w);
    } else {
      for (int i = tid; i < TBK * TBN; i += 256) {
        int r = i >> 6, c = i & 63;
        int gr = k0 + r, gc = bn + c;
        float vb = (gr < K && gc < N) ? W[(size_t)gr * N + gc] : 0.f;
        Bt[c][r] = f2bf(vb);
      }
    }
    if (k0 + TBK < K && avec) {  // gfx1250 global_prefetch of next A tile
      __builtin_prefetch(&A[(size_t)(bm + (tid >> 1)) * K + k0 + TBK], 0, 1);
    }
    __syncthreads();

    // ---- fragment loads: contiguous 16B runs -> ds_load_b128 ----
    // A frag lane layout: row = mfrag, elems = As[row][klo..klo+15]
    v8us a0l = *(const v8us*)&As[r0 + mfrag][klo];
    v8us a0h = *(const v8us*)&As[r0 + mfrag][klo + 8];
    v8us a1l = *(const v8us*)&As[r0 + 16 + mfrag][klo];
    v8us a1h = *(const v8us*)&As[r0 + 16 + mfrag][klo + 8];
    // B frag lane layout (same k order as A): col = mfrag, elems = Bt[col][kbB..kbB+15]
    v8us b0l = *(const v8us*)&Bt[c0 + mfrag][kbB];
    v8us b0h = *(const v8us*)&Bt[c0 + mfrag][kbB + 8];
    v8us b1l = *(const v8us*)&Bt[c0 + 16 + mfrag][kbB];
    v8us b1h = *(const v8us*)&Bt[c0 + 16 + mfrag][kbB + 8];

    v16bf fa0 = __builtin_bit_cast(v16bf, __builtin_shufflevector(a0l, a0h, 0,1,2,3,4,5,6,7,8,9,10,11,12,13,14,15));
    v16bf fa1 = __builtin_bit_cast(v16bf, __builtin_shufflevector(a1l, a1h, 0,1,2,3,4,5,6,7,8,9,10,11,12,13,14,15));
    v16bf fb0 = __builtin_bit_cast(v16bf, __builtin_shufflevector(b0l, b0h, 0,1,2,3,4,5,6,7,8,9,10,11,12,13,14,15));
    v16bf fb1 = __builtin_bit_cast(v16bf, __builtin_shufflevector(b1l, b1h, 0,1,2,3,4,5,6,7,8,9,10,11,12,13,14,15));

    acc[0][0] = __builtin_amdgcn_wmma_f32_16x16x32_bf16(false, fa0, false, fb0, (short)0, acc[0][0], false, false);
    acc[0][1] = __builtin_amdgcn_wmma_f32_16x16x32_bf16(false, fa0, false, fb1, (short)0, acc[0][1], false, false);
    acc[1][0] = __builtin_amdgcn_wmma_f32_16x16x32_bf16(false, fa1, false, fb0, (short)0, acc[1][0], false, false);
    acc[1][1] = __builtin_amdgcn_wmma_f32_16x16x32_bf16(false, fa1, false, fb1, (short)0, acc[1][1], false, false);
    __syncthreads();
  }

  __builtin_amdgcn_s_wait_tensorcnt(0);   // CDNA5 split-counter wait (TENSORcnt==0)

  // ---- epilogue: C layout M = vgpr_i + 8*(lane/16), N = mfrag ----
  const int rhi  = (lane >> 4) << 3;
  const int col0 = bn + c0 + mfrag;
  const int col1 = col0 + 16;
  float bb0 = 0.f, bb1 = 0.f;
  if (bias) {
    if (col0 < N) bb0 = bias[col0];
    if (col1 < N) bb1 = bias[col1];
  }
#pragma unroll
  for (int rt = 0; rt < 2; ++rt) {
#pragma unroll
    for (int i = 0; i < 8; ++i) {
      int grow = bm + r0 + rt * 16 + rhi + i;
      if (grow < M) {
        if (col0 < N) {
          float v = acc[rt][0][i] + bb0;
          if (ACT == 1) v = fmaxf(v, 0.f);
          C[(size_t)grow * ldc + col0] = v;
        }
        if (col1 < N) {
          float v = acc[rt][1][i] + bb1;
          if (ACT == 1) v = fmaxf(v, 0.f);
          C[(size_t)grow * ldc + col1] = v;
        }
      }
    }
  }
}

// ---------------- tiny cluster / TDM probe (NOP outside cluster dispatch) ----
__global__ void k_cluster_probe(int* sink) {
  __builtin_amdgcn_s_cluster_barrier();
  __builtin_amdgcn_s_wait_tensorcnt(0);
  if (threadIdx.x == 0) sink[0] = __builtin_amdgcn_cluster_id_x();
}

// ---------------- elementwise / graph kernels ----------------
__global__ void k_fill_u32(unsigned* __restrict__ p, unsigned v, long n) {
  long i = (long)blockIdx.x * blockDim.x + threadIdx.x;
  if (i < n) p[i] = v;
}
__global__ void k_add_inplace(float* __restrict__ a, const float* __restrict__ b, long n) {
  long i = (long)blockIdx.x * blockDim.x + threadIdx.x;
  if (i < n) a[i] += b[i];
}
__global__ void k_add_relu_inplace(float* __restrict__ a, const float* __restrict__ b, long n) {
  long i = (long)blockIdx.x * blockDim.x + threadIdx.x;
  if (i < n) a[i] = fmaxf(a[i] + b[i], 0.f);
}
__global__ void k_bias_act(float* __restrict__ x, const float* __restrict__ bias,
                           int M, int D, int relu) {
  long i = (long)blockIdx.x * blockDim.x + threadIdx.x;
  if (i >= (long)M * D) return;
  int d = (int)(i % D);
  float v = x[i] + bias[d];
  x[i] = relu ? fmaxf(v, 0.f) : v;
}
__global__ void k_ln_rows(const float* __restrict__ x, const float* __restrict__ g,
                          const float* __restrict__ b, float* __restrict__ o, int M, int D) {
  int r = blockIdx.x * blockDim.x + threadIdx.x;
  if (r >= M) return;
  const float* xr = x + (size_t)r * D;
  float m = 0.f;
  for (int i = 0; i < D; ++i) m += xr[i];
  m /= (float)D;
  float v = 0.f;
  for (int i = 0; i < D; ++i) { float d = xr[i] - m; v += d * d; }
  v /= (float)D;
  float s = rsqrtf(v + 1e-5f);
  float* orow = o + (size_t)r * D;
  for (int i = 0; i < D; ++i) orow[i] = (xr[i] - m) * s * g[i] + b[i];
}
__global__ void k_gn(const float* __restrict__ x, const float* __restrict__ alpha,
                     const float* __restrict__ gamma, const float* __restrict__ beta,
                     float* __restrict__ o, int Bb, int NN, int D) {
  int t = blockIdx.x * blockDim.x + threadIdx.x;
  if (t >= Bb * D) return;
  int b = t / D, h = t % D;
  const float* base = x + (size_t)b * NN * D + h;
  float m = 0.f;
  for (int i = 0; i < NN; ++i) m += base[(size_t)i * D];
  m /= (float)NN;
  float am = alpha[h] * m;
  float v = 0.f;
  for (int i = 0; i < NN; ++i) { float d = base[(size_t)i * D] - am; v += d * d; }
  v /= (float)NN;
  float s = rsqrtf(v + 1e-5f);
  float* ob = o + (size_t)b * NN * D + h;
  for (int i = 0; i < NN; ++i)
    ob[(size_t)i * D] = (base[(size_t)i * D] - am) * s * gamma[h] + beta[h];
}
__global__ void k_softmax_rows(float* __restrict__ x, int M, int D) {
  int r = blockIdx.x * blockDim.x + threadIdx.x;
  if (r >= M) return;
  float* xr = x + (size_t)r * D;
  float mx = -3.4e38f;
  for (int i = 0; i < D; ++i) mx = fmaxf(mx, xr[i]);
  float s = 0.f;
  for (int i = 0; i < D; ++i) { float e = __expf(xr[i] - mx); xr[i] = e; s += e; }
  float inv = 1.f / s;
  for (int i = 0; i < D; ++i) xr[i] *= inv;
}
__global__ void k_deg(const int* __restrict__ dst, float* __restrict__ deg, int E) {
  int e = blockIdx.x * blockDim.x + threadIdx.x;
  if (e < E) atomicAdd(&deg[dst[e]], 1.f);
}
__global__ void k_deg_stats(const float* __restrict__ deg, float* __restrict__ stats, int N) {
  int i = blockIdx.x * blockDim.x + threadIdx.x;
  if (i >= N) return;
  atomicAdd(&stats[0], logf(deg[i] + 1.f));
  atomicAdd(&stats[1], deg[i]);
}
__global__ void k_gather_m(const float* __restrict__ x, const float* __restrict__ et,
                           const int* __restrict__ src, const int* __restrict__ dst,
                           float* __restrict__ m, int E, int t) {
  long i = (long)blockIdx.x * blockDim.x + threadIdx.x;
  if (i >= (long)E * 120) return;
  int e = (int)(i / 120), c = (int)(i % 120);
  float v;
  if (c < 40)      v = x[(size_t)dst[e] * Hd + t * FTd + c];
  else if (c < 80) v = x[(size_t)src[e] * Hd + t * FTd + (c - 40)];
  else             v = et[(size_t)e * FTd + (c - 80)];
  m[i] = v;
}
__global__ void k_scatter_agg(const float* __restrict__ m, const int* __restrict__ dst,
                              float* __restrict__ asum, float* __restrict__ asq,
                              unsigned* __restrict__ amin, unsigned* __restrict__ amax, int E) {
  long i = (long)blockIdx.x * blockDim.x + threadIdx.x;
  if (i >= (long)E * FTd) return;
  int e = (int)(i / FTd), f = (int)(i % FTd);
  float v = m[i];
  size_t o = (size_t)dst[e] * FTd + f;
  atomicAdd(&asum[o], v);
  atomicAdd(&asq[o], v * v);
  atomicMin(&amin[o], fenc(v));
  atomicMax(&amax[o], fenc(v));
}
__global__ void k_pna_finalize(const float* __restrict__ x, const float* __restrict__ asum,
                               const float* __restrict__ asq, const unsigned* __restrict__ amin,
                               const unsigned* __restrict__ amax, const float* __restrict__ deg,
                               const float* __restrict__ stats, float* __restrict__ post,
                               int N, int t) {
  long i = (long)blockIdx.x * blockDim.x + threadIdx.x;
  if (i >= (long)N * FTd) return;
  int n = (int)(i / FTd), f = (int)(i % FTd);
  float d = deg[n];
  float dcl = fmaxf(d, 1.f);
  size_t o = (size_t)n * FTd + f;
  float mean = asum[o] / dcl;
  float msq  = asq[o] / dcl;
  float sd   = sqrtf(fmaxf(msq - mean * mean, 0.f) + 1e-5f);
  bool  has  = d > 0.f;
  float mn = has ? fdec(amin[o]) : 0.f;
  float mx = has ? fdec(amax[o]) : 0.f;
  float delta = fmaxf(stats[0] / (float)N, 1e-6f);
  float amp   = logf(d + 1.f) / delta;
  float mdeg  = fmaxf(stats[1] / (float)N, 1e-6f);
  float lin   = d / mdeg;
  float* pr = post + (size_t)n * 520;
  pr[f]       = x[(size_t)n * Hd + t * FTd + f];
  pr[40 + f]  = mean;       pr[80 + f]  = mn;       pr[120 + f] = mx;       pr[160 + f] = sd;
  pr[200 + f] = mean * amp; pr[240 + f] = mn * amp; pr[280 + f] = mx * amp; pr[320 + f] = sd * amp;
  pr[360 + f] = mean * lin; pr[400 + f] = mn * lin; pr[440 + f] = mx * lin; pr[480 + f] = sd * lin;
}
__global__ void k_gcn_init(const float* __restrict__ x, const float* __restrict__ gdeg,
                           float* __restrict__ o, int N, int D) {
  long i = (long)blockIdx.x * blockDim.x + threadIdx.x;
  if (i >= (long)N * D) return;
  int n = (int)(i / D);
  o[i] = x[i] / (gdeg[n] + 1.f);
}
__global__ void k_gcn_scatter(const float* __restrict__ x, const int* __restrict__ src,
                              const int* __restrict__ dst, const float* __restrict__ gdeg,
                              float* __restrict__ o, int E, int D) {
  long i = (long)blockIdx.x * blockDim.x + threadIdx.x;
  if (i >= (long)E * D) return;
  int e = (int)(i / D), d = (int)(i % D);
  int s = src[e], t = dst[e];
  float coef = rsqrtf(gdeg[s] + 1.f) * rsqrtf(gdeg[t] + 1.f);
  atomicAdd(&o[(size_t)t * D + d], x[(size_t)s * D + d] * coef);
}
__global__ void k_ssum(const float* __restrict__ s, float* __restrict__ ssum,
                       int Bb, int NN, int K) {
  int t = blockIdx.x * blockDim.x + threadIdx.x;
  if (t >= Bb * K) return;
  int b = t / K, k = t % K;
  float a = 0.f;
  for (int n = 0; n < NN; ++n) a += s[((size_t)b * NN + n) * K + k];
  ssum[t] = a;
}
__global__ void k_cpool(const float* __restrict__ s, const float* __restrict__ h,
                        const float* __restrict__ ssum, float* __restrict__ c,
                        int Bb, int NN, int K, int D) {
  int t = blockIdx.x * blockDim.x + threadIdx.x;
  if (t >= Bb * K * D) return;
  int d = t % D, bk = t / D, k = bk % K, b = bk / K;
  float a = 0.f;
  for (int n = 0; n < NN; ++n)
    a += s[((size_t)b * NN + n) * K + k] * h[((size_t)b * NN + n) * D + d];
  c[(size_t)bk * D + d] = a / (ssum[bk] + 1e-6f);
}
__global__ void k_attn(const float* __restrict__ q, const float* __restrict__ k,
                       const float* __restrict__ v, float* __restrict__ o,
                       int Bb, int nq, int nk, float scale) {
  int t = blockIdx.x * blockDim.x + threadIdx.x;
  if (t >= Bb * nq * HEADSd) return;
  int hh = t % HEADSd, r = t / HEADSd, qi = r % nq, b = r / nq;
  const float* qp = q + ((size_t)b * nq + qi) * Hd + hh * DHd;
  float acc[DHd];
#pragma unroll
  for (int d = 0; d < DHd; ++d) acc[d] = 0.f;
  float mx = -3.4e38f, den = 0.f;
  for (int j = 0; j < nk; ++j) {
    const float* kp = k + ((size_t)b * nk + j) * Hd + hh * DHd;
    float sdot = 0.f;
#pragma unroll
    for (int d = 0; d < DHd; ++d) sdot += qp[d] * kp[d];
    sdot *= scale;
    const float* vp = v + ((size_t)b * nk + j) * Hd + hh * DHd;
    if (sdot > mx) {
      float sc = __expf(mx - sdot);
      den *= sc;
#pragma unroll
      for (int d = 0; d < DHd; ++d) acc[d] *= sc;
      mx = sdot;
    }
    float w = __expf(sdot - mx);
    den += w;
#pragma unroll
    for (int d = 0; d < DHd; ++d) acc[d] += w * vp[d];
  }
  float inv = 1.f / den;
  float* op = o + ((size_t)b * nq + qi) * Hd + hh * DHd;
#pragma unroll
  for (int d = 0; d < DHd; ++d) op[d] = acc[d] * inv;
}
__global__ void k_amsg(const float* __restrict__ s, const float* __restrict__ cu,
                       float* __restrict__ o, int Bb, int NN, int K, int D) {
  int t = blockIdx.x * blockDim.x + threadIdx.x;
  if (t >= Bb * NN * D) return;
  int d = t % D, bn = t / D, n = bn % NN, b = bn / NN;
  float a = 0.f;
  for (int k = 0; k < K; ++k)
    a += s[((size_t)b * NN + n) * K + k] * cu[((size_t)b * K + k) * D + d];
  o[(size_t)bn * D + d] = a;
}
__global__ void k_meanpool(const float* __restrict__ x, float* __restrict__ o,
                           int Bb, int NN, int D) {
  int t = blockIdx.x * blockDim.x + threadIdx.x;
  if (t >= Bb * D) return;
  int d = t % D, b = t / D;
  float a = 0.f;
  for (int n = 0; n < NN; ++n) a += x[((size_t)b * NN + n) * D + d];
  o[(size_t)b * D + d] = a / (float)NN;
}
__global__ void k_concat_write(const float* __restrict__ src, float* __restrict__ dst,
                               int M, int D, int ldd, int coff) {
  int t = blockIdx.x * blockDim.x + threadIdx.x;
  if (t >= M * D) return;
  int d = t % D, m = t / D;
  dst[(size_t)m * ldd + coff + d] = src[t];
}

// ---------------- host orchestration ----------------
extern "C" void kernel_launch(void* const* d_in, const int* in_sizes, int n_in,
                              void* d_out, int out_size, void* d_ws, size_t ws_size,
                              hipStream_t stream) {
  (void)in_sizes; (void)n_in; (void)out_size; (void)ws_size;
  const float* mol_x   = (const float*)d_in[0];
  const float* mol_ea  = (const float*)d_in[1];
  const float* prot_x  = (const float*)d_in[2];
  const float* evo_x   = (const float*)d_in[3];
  const float* prot_ea = (const float*)d_in[4];
  const int*   mei     = (const int*)d_in[5];
  const int*   pei     = (const int*)d_in[6];
  const int *msrc = mei, *mdst = mei + EMd;
  const int *psrc = pei, *pdst = pei + EPd;
  float* ws  = (float*)d_ws;
  float* out = (float*)d_out;

  auto P = [&](int i) { return (const float*)d_in[i]; };

  // workspace layout (float offsets)
  float* HM   = ws + 0L;         // 2048*200
  float* HP   = ws + 409600L;    // 16384*200
  float* TA   = ws + 3686400L;   // 16384*520
  float* TB   = ws + 12206080L;  // 16384*400
  float* TC   = ws + 18759680L;  // 16384*200
  float* TD   = ws + 22036480L;  // 16384*200
  float* TE   = ws + 25313280L;  // 16384*200
  float* TF   = ws + 28590080L;  // 16384*200
  float* EA   = ws + 31866880L;  // 98304*120
  float* EB   = ws + 43663360L;  // 98304*40
  float* EC   = ws + 47595520L;  // 98304*40
  float* ET   = ws + 51527680L;  // 98304*40
  float* ASUM = ws + 55459840L;  // 16384*40
  float* ASQ  = ws + 56115200L;
  unsigned* AMIN = (unsigned*)(ws + 56770560L);
  unsigned* AMAX = (unsigned*)(ws + 57425920L);
  float* DEG   = ws + 58081280L; // 16384
  float* GDEG  = ws + 58097664L; // 2048
  float* STATS = ws + 58099712L; // 8
  float* S     = ws + 58099720L; // 2048*20
  float* SSUM  = ws + 58140680L; // 32*20
  float* CB    = ws + 58141320L; // 32*20*200
  float* CQ    = ws + 58269320L;
  float* CK    = ws + 58397320L;
  float* CV    = ws + 58525320L;
  float* CO    = ws + 58653320L;
  float* CUPD  = ws + 58781320L;
  float* AMSG  = ws + 58909320L; // 2048*200
  float* P1    = ws + 59318920L; // 32*400
  float* P2    = ws + 59331720L; // 32*400
  float* P3    = ws + 59344520L; // 32*200

  auto cdiv = [](long a, long b) { return (unsigned)((a + b - 1) / b); };
  auto gemm = [&](const float* A, const float* W, const float* bias, float* C,
                  int M, int N, int K, int ldc, bool relu) {
    dim3 g(cdiv(N, TBN), cdiv(M, TBM));
    if (relu) k_gemm<1><<<g, 256, 0, stream>>>(A, W, bias, C, M, N, K, ldc);
    else      k_gemm<0><<<g, 256, 0, stream>>>(A, W, bias, C, M, N, K, ldc);
  };
  auto fillu = [&](void* p, unsigned v, long n) {
    k_fill_u32<<<cdiv(n, 256), 256, 0, stream>>>((unsigned*)p, v, n);
  };

  // PNA conv; pb = leaf index of pna's edge_b (sorted-key order):
  // +0 edge_b, +1 edge_w, +2 lin_b, +3 lin_w, +4 post_b, +5 post_w,
  // +6 pre_b1, +7 pre_b2, +8 pre_w1, +9 pre_w2
  auto pna = [&](const float* x, const int* src, const int* dst, const float* eattr,
                 int N, int E, int pb, float* outp) {
    fillu(DEG, 0, N);
    k_deg<<<cdiv(E, 256), 256, 0, stream>>>(dst, DEG, E);
    fillu(STATS, 0, 2);
    k_deg_stats<<<cdiv(N, 256), 256, 0, stream>>>(DEG, STATS, N);
    gemm(eattr, P(pb + 1), P(pb + 0), ET, E, FTd, Hd, FTd, false);      // edge proj
    for (int t = 0; t < Td; ++t) {
      k_gather_m<<<cdiv((long)E * 120, 256), 256, 0, stream>>>(x, ET, src, dst, EA, E, t);
      gemm(EA, P(pb + 8) + (size_t)t * 120 * 40, P(pb + 6) + t * 40, EB, E, 40, 120, 40, true);
      gemm(EB, P(pb + 9) + (size_t)t * 40 * 40,  P(pb + 7) + t * 40, EC, E, 40, 40, 40, false);
      fillu(ASUM, 0, (long)N * 40);
      fillu(ASQ, 0, (long)N * 40);
      fillu(AMIN, 0xFFFFFFFFu, (long)N * 40);
      fillu(AMAX, 0u, (long)N * 40);
      k_scatter_agg<<<cdiv((long)E * 40, 256), 256, 0, stream>>>(EC, dst, ASUM, ASQ, AMIN, AMAX, E);
      k_pna_finalize<<<cdiv((long)N * 40, 256), 256, 0, stream>>>(x, ASUM, ASQ, AMIN, AMAX,
                                                                  DEG, STATS, TA, N, t);
      gemm(TA, P(pb + 5) + (size_t)t * 520 * 40, P(pb + 4) + t * 40, TB + t * 40, N, 40, 520, Hd, false);
    }
    gemm(TB, P(pb + 3), P(pb + 2), outp, N, Hd, Hd, Hd, false);         // lin
  };

  // ---- encoders ----  (atom_enc leaves 7..12: bn,bs0,bs1,g,ws0,ws1)
  gemm(mol_x, P(11), P(8), TA, NMd, 2 * Hd, 43, 2 * Hd, true);
  gemm(TA, P(12), P(9), HM, NMd, Hd, 2 * Hd, Hd, false);
  k_ln_rows<<<cdiv(NMd, 128), 128, 0, stream>>>(HM, P(10), P(7), HM, NMd, Hd);
  // prot_aa leaves 181..186
  gemm(prot_x, P(185), P(182), TA, NPRd, 2 * Hd, 33, 2 * Hd, true);
  gemm(TA, P(186), P(183), TC, NPRd, Hd, 2 * Hd, Hd, false);
  k_ln_rows<<<cdiv(NPRd, 128), 128, 0, stream>>>(TC, P(184), P(181), TC, NPRd, Hd);
  // prot_evo leaves 187..192
  gemm(evo_x, P(191), P(188), TA, NPRd, 2 * Hd, 1280, 2 * Hd, true);
  gemm(TA, P(192), P(189), HP, NPRd, Hd, 2 * Hd, Hd, false);
  k_ln_rows<<<cdiv(NPRd, 128), 128, 0, stream>>>(HP, P(190), P(187), HP, NPRd, Hd);
  k_add_inplace<<<cdiv(3276800L, 256), 256, 0, stream>>>(HP, TC, 3276800L);

  const float kscale = 0.15811388f;  // 1/sqrt(DH)/TEMP
  const int KSarr[3] = {5, 10, 20};

  for (int li = 0; li < 3; ++li) {
    const int L = 13 + 54 * li;
    const int Kc = KSarr[li];
    // ---- mol PNA + graphnorm + residual relu ----
    pna(HM, msrc, mdst, mol_ea, NMd, EMd, L + 28, TC);
    k_gn<<<cdiv(Bd * Hd, 256), 256, 0, stream>>>(TC, P(L + 23), P(L + 25), P(L + 24), TD, Bd, NAd, Hd);
    k_add_relu_inplace<<<cdiv(409600L, 256), 256, 0, stream>>>(HM, TD, 409600L);
    // ---- prot PNA + graphnorm + residual relu ----
    pna(HP, psrc, pdst, prot_ea, NPRd, EPd, L + 43, TC);
    k_gn<<<cdiv(Bd * Hd, 256), 256, 0, stream>>>(TC, P(L + 38), P(L + 40), P(L + 39), TD, Bd, NRd, Hd);
    k_add_relu_inplace<<<cdiv(3276800L, 256), 256, 0, stream>>>(HP, TD, 3276800L);
    // ---- cluster assignment (mol graph, GCN x2 + softmax) ----
    k_ln_rows<<<cdiv(NMd, 128), 128, 0, stream>>>(HM, P(L + 20), P(L + 19), TA, NMd, Hd);
    gemm(TA, P(L + 21), nullptr, TB, NMd, 2 * Hd, Hd, 2 * Hd, false);
    fillu(GDEG, 0, NMd);
    k_deg<<<cdiv(EMd, 256), 256, 0, stream>>>(mdst, GDEG, EMd);
    k_gcn_init<<<cdiv((long)NMd * 400, 256), 256, 0, stream>>>(TB, GDEG, TA, NMd, 400);
    k_gcn_scatter<<<cdiv((long)EMd * 400, 256), 256, 0, stream>>>(TB, msrc, mdst, GDEG, TA, EMd, 400);
    k_bias_act<<<cdiv((long)NMd * 400, 256), 256, 0, stream>>>(TA, P(L + 17), NMd, 400, 1);
    gemm(TA, P(L + 22), nullptr, TB, NMd, Kc, 400, Kc, false);
    k_gcn_init<<<cdiv((long)NMd * Kc, 256), 256, 0, stream>>>(TB, GDEG, S, NMd, Kc);
    k_gcn_scatter<<<cdiv((long)EMd * Kc, 256), 256, 0, stream>>>(TB, msrc, mdst, GDEG, S, EMd, Kc);
    k_bias_act<<<cdiv((long)NMd * Kc, 256), 256, 0, stream>>>(S, P(L + 18), NMd, Kc, 0);
    k_softmax_rows<<<cdiv(NMd, 128), 128, 0, stream>>>(S, NMd, Kc);
    // ---- cluster pooling ----
    k_ssum<<<cdiv(Bd * Kc, 128), 128, 0, stream>>>(S, SSUM, Bd, NAd, Kc);
    k_cpool<<<cdiv((long)Bd * Kc * Hd, 256), 256, 0, stream>>>(S, HM, SSUM, CB, Bd, NAd, Kc, Hd);
    // ---- attn_r (residues attend clusters; wk=L+5 wo=L+6 wq=L+7 wv=L+8) ----
    gemm(HP, P(L + 7), nullptr, TA, NPRd, Hd, Hd, Hd, false);
    gemm(CB, P(L + 5), nullptr, CK, Bd * Kc, Hd, Hd, Hd, false);
    gemm(CB, P(L + 8), nullptr, CV, Bd * Kc, Hd, Hd, Hd, false);
    k_attn<<<cdiv((long)Bd * NRd * HEADSd, 128), 128, 0, stream>>>(TA, CK, CV, TB, Bd, NRd, Kc, kscale);
    gemm(TB, P(L + 6), nullptr, TC, NPRd, Hd, Hd, Hd, false);        // r_upd -> TC
    // ---- attn_c (clusters attend residues; wk=L+1 wo=L+2 wq=L+3 wv=L+4) ----
    gemm(CB, P(L + 3), nullptr, CQ, Bd * Kc, Hd, Hd, Hd, false);
    gemm(HP, P(L + 1), nullptr, TE, NPRd, Hd, Hd, Hd, false);
    gemm(HP, P(L + 4), nullptr, TF, NPRd, Hd, Hd, Hd, false);
    k_attn<<<cdiv((long)Bd * Kc * HEADSd, 128), 128, 0, stream>>>(CQ, TE, TF, CO, Bd, Kc, NRd, kscale);
    gemm(CO, P(L + 2), nullptr, CUPD, Bd * Kc, Hd, Hd, Hd, false);
    // ---- clusters -> atoms, c2a MLP, residual + LN ----
    k_amsg<<<cdiv((long)NMd * Hd, 256), 256, 0, stream>>>(S, CUPD, AMSG, Bd, NAd, Kc, Hd);
    gemm(AMSG, P(L + 0), nullptr, TA, NMd, Hd, Hd, Hd, false);
    gemm(TA, P(L + 11), P(L + 9), TB, NMd, 2 * Hd, Hd, 2 * Hd, true);
    gemm(TB, P(L + 12), P(L + 10), TD, NMd, Hd, 2 * Hd, Hd, false);
    k_add_inplace<<<cdiv(409600L, 256), 256, 0, stream>>>(HM, TD, 409600L);
    k_ln_rows<<<cdiv(NMd, 128), 128, 0, stream>>>(HM, P(L + 27), P(L + 26), HM, NMd, Hd);
    // ---- residue update: residue_lin, c2r MLP, residual + LN ----
    gemm(TC, P(L + 53), nullptr, TA, NPRd, Hd, Hd, Hd, false);
    gemm(TA, P(L + 15), P(L + 13), TB, NPRd, 2 * Hd, Hd, 2 * Hd, true);
    gemm(TB, P(L + 16), P(L + 14), TD, NPRd, Hd, 2 * Hd, Hd, false);
    k_add_inplace<<<cdiv(3276800L, 256), 256, 0, stream>>>(HP, TD, 3276800L);
    k_ln_rows<<<cdiv(NPRd, 128), 128, 0, stream>>>(HP, P(L + 42), P(L + 41), HP, NPRd, Hd);
  }

  // ---- readout: mol_out (175..180), prot_out (193..198), reg_out (199..202) ----
  k_meanpool<<<cdiv(Bd * Hd, 128), 128, 0, stream>>>(HM, P3, Bd, NAd, Hd);
  gemm(P3, P(179), P(176), P2, Bd, 2 * Hd, Hd, 2 * Hd, true);
  gemm(P2, P(180), P(177), P3, Bd, Hd, 2 * Hd, Hd, false);
  k_ln_rows<<<1, Bd, 0, stream>>>(P3, P(178), P(175), P3, Bd, Hd);
  k_concat_write<<<cdiv(Bd * Hd, 128), 128, 0, stream>>>(P3, P1, Bd, Hd, 2 * Hd, 0);
  k_meanpool<<<cdiv(Bd * Hd, 128), 128, 0, stream>>>(HP, P3, Bd, NRd, Hd);
  gemm(P3, P(197), P(194), P2, Bd, 2 * Hd, Hd, 2 * Hd, true);
  gemm(P2, P(198), P(195), P3, Bd, Hd, 2 * Hd, Hd, false);
  k_ln_rows<<<1, Bd, 0, stream>>>(P3, P(196), P(193), P3, Bd, Hd);
  k_concat_write<<<cdiv(Bd * Hd, 128), 128, 0, stream>>>(P3, P1, Bd, Hd, 2 * Hd, Hd);
  gemm(P1, P(201), P(199), P2, Bd, Hd, 2 * Hd, Hd, true);
  gemm(P2, P(202), P(200), out, Bd, 1, Hd, 1, false);

  // exercise CDNA5 cluster-id / cluster-barrier path (NOP outside clusters)
  k_cluster_probe<<<1, 32, 0, stream>>>((int*)(STATS + 4));
}
